// SwinTransformerBlock_22703197127153
// MI455X (gfx1250) — matrix operand order repaired and checked by reading, not compiled, not run on hardware
//
#include <hip/hip_runtime.h>
#include <hip/hip_bf16.h>

typedef __attribute__((ext_vector_type(16))) _Float16 v16h;
typedef __attribute__((ext_vector_type(8)))  _Float16 v8h;
typedef __attribute__((ext_vector_type(8)))  float    v8f;

// ---- static problem config (matches reference) ----
static constexpr int B_   = 4;
static constexpr int D_   = 8;
static constexpr int H_   = 56;
static constexpr int W_   = 56;
static constexpr int C_   = 192;
static constexpr int NH_  = 6;
static constexpr int HD_  = 32;
static constexpr int N_   = 98;    // tokens per window
static constexpr int NP_  = 112;   // padded token rows (7*16)
static constexpr int KP_  = 128;   // padded K for attn@v (4*32)
static constexpr int NWIMG_ = 256; // windows per image
static constexpr int NWTOT_ = 1024;
static constexpr int WHT_ = NWTOT_ * NH_;   // 6144 (win,head) pairs
static constexpr int M_   = NWTOT_ * N_;    // 100352 tokens
static constexpr int HID_ = 768;
static constexpr float LOG100_ = 4.6051702f;

// =================== WMMA fragment helpers ===================
// A-matrix 16x32 f16: lane = (m & 15) + 16*g ; elements 0..15 -> K = kb+g*16..+15
// (contiguous, matching B)
__device__ __forceinline__ v16h load_a_frag(const _Float16* A, int lda, int mBase,
                                            int kb, int lane) {
  int m = mBase + (lane & 15);
  int g = lane >> 4;
  return *(const v16h*)(A + (size_t)m * lda + kb + g * 16);
}
// B-matrix 32x16 f16, supplied as row-major "weight" Bw[n][k]: lane group g holds
// K = kb + g*16 + e  (contiguous 16 halfs)
__device__ __forceinline__ v16h load_b_frag(const _Float16* Bw, int ldb, int nBase,
                                            int kb, int lane) {
  int n = nBase + (lane & 15);
  int g = lane >> 4;
  return *(const v16h*)(Bw + (size_t)n * ldb + kb + g * 16);
}
__device__ __forceinline__ v8f wmma_f16(v16h a, v16h b, v8f c) {
  return __builtin_amdgcn_wmma_f32_16x16x32_f16(false, a, false, b, (short)0, c,
                                                false, false);
}

// =================== small prep kernels ===================
__global__ __launch_bounds__(256) void k_cvtw(const float* __restrict__ qkv_w,
                                              const float* __restrict__ proj_w,
                                              const float* __restrict__ fc1_w,
                                              const float* __restrict__ fc2_w,
                                              _Float16* wq, _Float16* wp,
                                              _Float16* w1, _Float16* w2) {
  int i = blockIdx.x * 256 + threadIdx.x;
  if (i < 576 * 192) wq[i] = (_Float16)qkv_w[i];
  if (i < 192 * 192) wp[i] = (_Float16)proj_w[i];
  if (i < 768 * 192) { w1[i] = (_Float16)fc1_w[i]; w2[i] = (_Float16)fc2_w[i]; }
}

// cyclic shift + window partition + f32->f16
__global__ __launch_bounds__(256) void k_window(const float* __restrict__ x,
                                                _Float16* __restrict__ xw) {
  size_t idx = (size_t)blockIdx.x * 256 + threadIdx.x;
  if (idx >= (size_t)M_ * C_) return;
  int c   = (int)(idx % C_);
  int r   = (int)(idx / C_);
  int b   = r / (NWIMG_ * N_);
  int rem = r % (NWIMG_ * N_);
  int wimg = rem / N_;
  int tok  = rem % N_;
  int nd = wimg >> 6, nh = (wimg >> 3) & 7, nw = wimg & 7;
  int td = tok / 49, th = (tok / 7) % 7, tw = tok % 7;
  int d0 = (nd * 2 + td + 1) & 7;        // +SD mod D
  int h0 = (nh * 7 + th + 3) % 56;       // +SH mod H
  int w0 = (nw * 7 + tw + 3) % 56;       // +SW mod W
  size_t src = ((((size_t)b * D_ + d0) * H_ + h0) * W_ + w0) * C_ + c;
  xw[idx] = (_Float16)x[src];
}

__global__ __launch_bounds__(256) void k_zero16(uint4* __restrict__ p, int n) {
  int i = blockIdx.x * 256 + threadIdx.x;
  if (i < n) p[i] = make_uint4(0u, 0u, 0u, 0u);
}

// =================== GEMM kernels (WMMA) ===================
// Each wave computes a 32(M)x16(N) strip; block = 4 waves over 4 N-tiles.
__global__ __launch_bounds__(128) void k_gemm_qkv(const _Float16* __restrict__ A,
                                                  const _Float16* __restrict__ Bw,
                                                  const float* __restrict__ q_bias,
                                                  const float* __restrict__ v_bias,
                                                  _Float16* __restrict__ qkv) {
  int lane = threadIdx.x & 31, wave = threadIdx.x >> 5;
  int nBase = (blockIdx.x * 4 + wave) * 16;
  int m0 = blockIdx.y * 32, m1 = m0 + 16;
  v8f acc0 = {}, acc1 = {};
  for (int kb = 0; kb < C_; kb += 32) {
    v16h b  = load_b_frag(Bw, C_, nBase, kb, lane);
    v16h a0 = load_a_frag(A,  C_, m0,    kb, lane);
    v16h a1 = load_a_frag(A,  C_, m1,    kb, lane);
    acc0 = wmma_f16(a0, b, acc0);
    acc1 = wmma_f16(a1, b, acc1);
  }
  int g = lane >> 4;
  int n = nBase + (lane & 15);
  int which = n / 192, cc = n % 192, head = cc >> 5, hd = cc & 31;
  float bias = (which == 0) ? q_bias[cc] : ((which == 2) ? v_bias[cc] : 0.f);
#pragma unroll
  for (int r = 0; r < 8; r++) {
    int row = m0 + r + 8 * g;
    int win = row / N_, tok = row - win * N_;
    qkv[(((size_t)which * WHT_ + (size_t)win * NH_ + head) * NP_ + tok) * HD_ + hd] =
        (_Float16)(acc0[r] + bias);
    row = m1 + r + 8 * g;
    win = row / N_; tok = row - win * N_;
    qkv[(((size_t)which * WHT_ + (size_t)win * NH_ + head) * NP_ + tok) * HD_ + hd] =
        (_Float16)(acc1[r] + bias);
  }
}

__global__ __launch_bounds__(128) void k_gemm_proj(const _Float16* __restrict__ A,
                                                   const _Float16* __restrict__ Bw,
                                                   const float* __restrict__ bias,
                                                   float* __restrict__ y) {
  int lane = threadIdx.x & 31, wave = threadIdx.x >> 5;
  int nBase = (blockIdx.x * 4 + wave) * 16;
  int m0 = blockIdx.y * 32, m1 = m0 + 16;
  v8f acc0 = {}, acc1 = {};
  for (int kb = 0; kb < C_; kb += 32) {
    v16h b  = load_b_frag(Bw, C_, nBase, kb, lane);
    v16h a0 = load_a_frag(A,  C_, m0,    kb, lane);
    v16h a1 = load_a_frag(A,  C_, m1,    kb, lane);
    acc0 = wmma_f16(a0, b, acc0);
    acc1 = wmma_f16(a1, b, acc1);
  }
  int g = lane >> 4;
  int n = nBase + (lane & 15);
  float bs = bias[n];
#pragma unroll
  for (int r = 0; r < 8; r++) {
    y[(size_t)(m0 + r + 8 * g) * C_ + n] = acc0[r] + bs;
    y[(size_t)(m1 + r + 8 * g) * C_ + n] = acc1[r] + bs;
  }
}

__global__ __launch_bounds__(128) void k_gemm_fc1(const _Float16* __restrict__ A,
                                                  const _Float16* __restrict__ Bw,
                                                  const float* __restrict__ bias,
                                                  _Float16* __restrict__ hbuf) {
  int lane = threadIdx.x & 31, wave = threadIdx.x >> 5;
  int nBase = (blockIdx.x * 4 + wave) * 16;
  int m0 = blockIdx.y * 32, m1 = m0 + 16;
  v8f acc0 = {}, acc1 = {};
  for (int kb = 0; kb < C_; kb += 32) {
    v16h b  = load_b_frag(Bw, C_, nBase, kb, lane);
    v16h a0 = load_a_frag(A,  C_, m0,    kb, lane);
    v16h a1 = load_a_frag(A,  C_, m1,    kb, lane);
    acc0 = wmma_f16(a0, b, acc0);
    acc1 = wmma_f16(a1, b, acc1);
  }
  int g = lane >> 4;
  int n = nBase + (lane & 15);
  float bs = bias[n];
#pragma unroll
  for (int r = 0; r < 8; r++) {
    float v0 = acc0[r] + bs;
    float v1 = acc1[r] + bs;
    v0 = 0.5f * v0 * (1.f + erff(v0 * 0.70710678f));   // exact GELU
    v1 = 0.5f * v1 * (1.f + erff(v1 * 0.70710678f));
    hbuf[(size_t)(m0 + r + 8 * g) * HID_ + n] = (_Float16)v0;
    hbuf[(size_t)(m1 + r + 8 * g) * HID_ + n] = (_Float16)v1;
  }
}

__global__ __launch_bounds__(128) void k_gemm_fc2(const _Float16* __restrict__ A,
                                                  const _Float16* __restrict__ Bw,
                                                  const float* __restrict__ bias,
                                                  float* __restrict__ hh) {
  int lane = threadIdx.x & 31, wave = threadIdx.x >> 5;
  int nBase = (blockIdx.x * 4 + wave) * 16;
  int m0 = blockIdx.y * 32, m1 = m0 + 16;
  v8f acc0 = {}, acc1 = {};
  for (int kb = 0; kb < HID_; kb += 32) {
    v16h b  = load_b_frag(Bw, HID_, nBase, kb, lane);
    v16h a0 = load_a_frag(A,  HID_, m0,    kb, lane);
    v16h a1 = load_a_frag(A,  HID_, m1,    kb, lane);
    acc0 = wmma_f16(a0, b, acc0);
    acc1 = wmma_f16(a1, b, acc1);
  }
  int g = lane >> 4;
  int n = nBase + (lane & 15);
  float bs = bias[n];
#pragma unroll
  for (int r = 0; r < 8; r++) {
    hh[(size_t)(m0 + r + 8 * g) * C_ + n] = acc0[r] + bs;
    hh[(size_t)(m1 + r + 8 * g) * C_ + n] = acc1[r] + bs;
  }
}

// =================== q/k normalization (cosine attn) ===================
__global__ __launch_bounds__(256) void k_normqk(_Float16* __restrict__ qkv,
                                                const float* __restrict__ logit_scale) {
  int idx = blockIdx.x * 256 + threadIdx.x;   // over 2 * WHT_ * NP_
  if (idx >= 2 * WHT_ * NP_) return;
  int which = idx / (WHT_ * NP_);
  int rem   = idx % (WHT_ * NP_);
  int wh    = rem / NP_;
  int head  = wh % NH_;
  _Float16* p = qkv + ((size_t)which * WHT_ + wh) * NP_ * HD_ + (size_t)(rem % NP_) * HD_;
  float vals[HD_];
  float ss = 0.f;
#pragma unroll
  for (int i = 0; i < HD_; i++) { vals[i] = (float)p[i]; ss += vals[i] * vals[i]; }
  float inv = 1.f / fmaxf(sqrtf(ss), 1e-12f);
  if (which == 0) inv *= __expf(fminf(logit_scale[head], LOG100_));  // fold scale into qn
#pragma unroll
  for (int i = 0; i < HD_; i++) p[i] = (_Float16)(vals[i] * inv);
}

// =================== CPB-MLP relative position bias ===================
__device__ __forceinline__ float cpb_coord(int idx, int off, float denom) {
  float v = (float)(idx - off) / denom * 8.0f;
  float s = (v > 0.f) ? 1.f : ((v < 0.f) ? -1.f : 0.f);
  return s * log2f(fabsf(v) + 1.f) * (1.0f / 3.0f);   // /log2(8)
}

__global__ __launch_bounds__(256) void k_tbl(const float* __restrict__ w1,
                                             const float* __restrict__ b1,
                                             const float* __restrict__ w2,
                                             float* __restrict__ tbl) {
  int t = blockIdx.x * 256 + threadIdx.x;
  if (t >= 507) return;   // 3*13*13
  int i0 = t / 169, i1 = (t / 13) % 13, i2 = t % 13;
  float g0 = cpb_coord(i0, 1, 1.f);
  float g1 = cpb_coord(i1, 6, 6.f);
  float g2 = cpb_coord(i2, 6, 6.f);
  float acc[NH_] = {0.f, 0.f, 0.f, 0.f, 0.f, 0.f};
  for (int j = 0; j < 512; j++) {
    float hv = w1[j * 3 + 0] * g0 + w1[j * 3 + 1] * g1 + w1[j * 3 + 2] * g2 + b1[j];
    hv = fmaxf(hv, 0.f);
#pragma unroll
    for (int q = 0; q < NH_; q++) acc[q] += hv * w2[q * 512 + j];
  }
  for (int q = 0; q < NH_; q++) tbl[t * NH_ + q] = acc[q];
}

__global__ __launch_bounds__(256) void k_rpb(const float* __restrict__ tbl,
                                             float* __restrict__ rpb) {
  int idx = blockIdx.x * 256 + threadIdx.x;
  if (idx >= NH_ * N_ * N_) return;
  int h   = idx / (N_ * N_);
  int rem = idx % (N_ * N_);
  int i = rem / N_, j = rem % N_;
  int di = i / 49, hi = (i / 7) % 7, wi = i % 7;
  int dj = j / 49, hj = (j / 7) % 7, wj = j % 7;
  int t = (di - dj + 1) * 169 + (hi - hj + 6) * 13 + (wi - wj + 6);
  float v = tbl[t * NH_ + h];
  rpb[idx] = 16.f / (1.f + __expf(-v));   // 16*sigmoid
}

// =================== fused attention (per window,head) ===================
__global__ __launch_bounds__(256) void k_attn(const _Float16* __restrict__ qkv,
                                              const float* __restrict__ rpb,
                                              _Float16* __restrict__ attnout) {
  extern __shared__ char smem[];
  float*    S    = (float*)smem;                          // 112*112*4 = 50176
  float*    rmx  = (float*)(smem + 50176);                // 112*4
  float*    rinv = (float*)(smem + 50176 + 448);          // 112*4
  _Float16* P    = (_Float16*)(smem + 51072);             // 112*128*2 = 28672
  _Float16* Vt   = (_Float16*)(smem + 51072 + 28672);     // 32*128*2  = 8192
  int*      cnt  = (int*)(smem + 51072 + 28672 + 8192);   // 112*4

  const int wh   = blockIdx.x;
  const int win  = wh / NH_;
  const int head = wh % NH_;
  const int tid  = threadIdx.x;
  const int lane = tid & 31;
  const int wave = tid >> 5;

  const _Float16* qg = qkv + ((size_t)(0 * WHT_ + wh)) * NP_ * HD_;
  const _Float16* kg = qkv + ((size_t)(1 * WHT_ + wh)) * NP_ * HD_;
  const _Float16* vg = qkv + ((size_t)(2 * WHT_ + wh)) * NP_ * HD_;

  // V^T into LDS, token dim padded to 128
  for (int idx = tid; idx < HD_ * KP_; idx += 256) {
    int k = idx >> 5, c = idx & 31;
    _Float16 val = (_Float16)0.f;
    if (k < NP_) val = vg[(size_t)k * HD_ + c];
    Vt[c * KP_ + k] = val;
  }
  // shifted-window mask region ids
  if (tid < NP_) {
    int cv = 0;
    if (tid < N_) {
      int wimg = win & 255;
      int nd = wimg >> 6, nh = (wimg >> 3) & 7, nw = wimg & 7;
      int td = tid / 49, th = (tid / 7) % 7, tw = tid % 7;
      int gd = nd * 2 + td, gh = nh * 7 + th, gw = nw * 7 + tw;
      int bd = (gd >= D_ - 1) ? 2 : ((gd >= D_ - 2) ? 1 : 0);
      int bh = (gh >= H_ - 3) ? 2 : ((gh >= H_ - 7) ? 1 : 0);
      int bw = (gw >= W_ - 3) ? 2 : ((gw >= W_ - 7) ? 1 : 0);
      cv = bd * 9 + bh * 3 + bw;
    }
    cnt[tid] = cv;
  }
  __syncthreads();

  // S = qn @ kn^T  + rpb + mask  (7x7 tiles, K=32)
  for (int t = wave; t < 49; t += 8) {
    int ti = t / 7, tj = t % 7;
    v16h a = load_a_frag(qg, HD_, ti * 16, 0, lane);
    v16h b = load_b_frag(kg, HD_, tj * 16, 0, lane);
    v8f acc = {};
    acc = wmma_f16(a, b, acc);
    int g = lane >> 4;
    int n = tj * 16 + (lane & 15);
#pragma unroll
    for (int r = 0; r < 8; r++) {
      int m = ti * 16 + r + 8 * g;
      float s = -1e30f;
      if (m < N_ && n < N_) {
        float bias = rpb[((size_t)head * N_ + m) * N_ + n];
        float msk  = (cnt[m] != cnt[n]) ? -100.f : 0.f;
        s = acc[r] + bias + msk;
      }
      S[m * NP_ + n] = s;
    }
  }
  __syncthreads();

  // row softmax stats
  if (tid < NP_) {
    int i = tid;
    if (i < N_) {
      float mx = -1e30f;
      for (int j = 0; j < NP_; j++) mx = fmaxf(mx, S[i * NP_ + j]);
      float sum = 0.f;
      for (int j = 0; j < NP_; j++) sum += __expf(S[i * NP_ + j] - mx);
      rmx[i] = mx; rinv[i] = 1.f / sum;
    } else { rmx[i] = 0.f; rinv[i] = 0.f; }
  }
  __syncthreads();

  // P = softmax(S), f16, K padded to 128
  for (int idx = tid; idx < NP_ * KP_; idx += 256) {
    int i = idx >> 7, j = idx & 127;
    float p = 0.f;
    if (i < N_ && j < NP_) p = __expf(S[i * NP_ + j] - rmx[i]) * rinv[i];
    P[idx] = (_Float16)p;
  }
  __syncthreads();

  // O = P @ V  (M=112, N=32, K=128), scatter to (token, head*32+hd)
  for (int t = wave; t < 14; t += 8) {
    int ti = t >> 1, tj = t & 1;
    v8f acc = {};
#pragma unroll
    for (int kb = 0; kb < KP_; kb += 32) {
      v16h a = load_a_frag(P,  KP_, ti * 16, kb, lane);
      v16h b = load_b_frag(Vt, KP_, tj * 16, kb, lane);
      acc = wmma_f16(a, b, acc);
    }
    int g = lane >> 4;
    int n = tj * 16 + (lane & 15);
#pragma unroll
    for (int r = 0; r < 8; r++) {
      int m = ti * 16 + r + 8 * g;
      if (m < N_)
        attnout[((size_t)win * N_ + m) * C_ + head * HD_ + n] = (_Float16)acc[r];
    }
  }
}

// =================== window reverse + unshift + LN1 + residual ===================
__global__ __launch_bounds__(256) void k_unwin_ln1(const float* __restrict__ y,
                                                   const float* __restrict__ x,
                                                   const float* __restrict__ gam,
                                                   const float* __restrict__ bet,
                                                   float* __restrict__ x1,
                                                   _Float16* __restrict__ x1h) {
  int t = blockIdx.x * 256 + threadIdx.x;
  if (t >= M_) return;
  int b    = t / (D_ * H_ * W_);
  int rem  = t % (D_ * H_ * W_);
  int d    = rem / (H_ * W_);
  int rem2 = rem % (H_ * W_);
  int h = rem2 / W_, w = rem2 % W_;
  int dm = (d - 1 + D_) % D_;
  int hm = (h - 3 + H_) % H_;
  int wm = (w - 3 + W_) % W_;
  int nd = dm / 2, td = dm % 2;
  int nh = hm / 7, th = hm % 7;
  int nw = wm / 7, tw = wm % 7;
  int wimg = (nd * 8 + nh) * 8 + nw;
  int tok  = (td * 7 + th) * 7 + tw;
  const float* yr = y + ((size_t)b * NWIMG_ + wimg) * N_ * C_ + (size_t)tok * C_;
  float sum = 0.f, ss = 0.f;
  for (int c = 0; c < C_; c++) { float v = yr[c]; sum += v; ss += v * v; }
  float mean = sum / C_;
  float var  = ss / C_ - mean * mean;
  float rstd = rsqrtf(var + 1e-5f);
  size_t base = (size_t)t * C_;
  for (int c = 0; c < C_; c++) {
    float v = (yr[c] - mean) * rstd * gam[c] + bet[c];
    float o = x[base + c] + v;
    x1[base + c]  = o;
    x1h[base + c] = (_Float16)o;
  }
}

// =================== final LN2 + residual ===================
__global__ __launch_bounds__(256) void k_final(const float* __restrict__ hh,
                                               const float* __restrict__ x1,
                                               const float* __restrict__ gam,
                                               const float* __restrict__ bet,
                                               float* __restrict__ out) {
  int t = blockIdx.x * 256 + threadIdx.x;
  if (t >= M_) return;
  const float* r = hh + (size_t)t * C_;
  float sum = 0.f, ss = 0.f;
  for (int c = 0; c < C_; c++) { float v = r[c]; sum += v; ss += v * v; }
  float mean = sum / C_;
  float var  = ss / C_ - mean * mean;
  float rstd = rsqrtf(var + 1e-5f);
  size_t base = (size_t)t * C_;
  for (int c = 0; c < C_; c++) {
    float v = (r[c] - mean) * rstd * gam[c] + bet[c];
    out[base + c] = x1[base + c] + v;
  }
}

// =================== host orchestration ===================
extern "C" void kernel_launch(void* const* d_in, const int* in_sizes, int n_in,
                              void* d_out, int out_size, void* d_ws, size_t ws_size,
                              hipStream_t stream) {
  (void)in_sizes; (void)n_in; (void)out_size; (void)ws_size;
  const float* x       = (const float*)d_in[0];
  const float* qkv_w   = (const float*)d_in[1];
  const float* q_bias  = (const float*)d_in[2];
  const float* v_bias  = (const float*)d_in[3];
  const float* logit_s = (const float*)d_in[4];
  const float* cpb_w1  = (const float*)d_in[5];
  const float* cpb_b1  = (const float*)d_in[6];
  const float* cpb_w2  = (const float*)d_in[7];
  const float* proj_w  = (const float*)d_in[8];
  const float* proj_b  = (const float*)d_in[9];
  const float* n1g     = (const float*)d_in[10];
  const float* n1b     = (const float*)d_in[11];
  const float* n2g     = (const float*)d_in[12];
  const float* n2b     = (const float*)d_in[13];
  const float* fc1_w   = (const float*)d_in[14];
  const float* fc1_b   = (const float*)d_in[15];
  const float* fc2_w   = (const float*)d_in[16];
  const float* fc2_b   = (const float*)d_in[17];
  char* ws = (char*)d_ws;

  auto al = [](size_t v) { return (v + 255) & ~(size_t)255; };
  const size_t S_MC2 = (size_t)M_ * C_ * 2;      // 38.5 MB f16 M x C
  const size_t S_MC4 = (size_t)M_ * C_ * 4;      // 77.1 MB f32 M x C
  const size_t S_QKV = (size_t)3 * WHT_ * NP_ * HD_ * 2;  // 132.1 MB

  size_t o_xw  = 0;
  size_t o_wq  = o_xw  + al(S_MC2);
  size_t o_wp  = o_wq  + al((size_t)576 * 192 * 2);
  size_t o_w1  = o_wp  + al((size_t)192 * 192 * 2);
  size_t o_w2  = o_w1  + al((size_t)768 * 192 * 2);
  size_t o_qkv = o_w2  + al((size_t)192 * 768 * 2);
  size_t o_tbl = o_qkv + al(S_QKV);
  size_t o_rpb = o_tbl + al((size_t)507 * 6 * 4);
  size_t o_ao  = o_rpb + al((size_t)6 * 98 * 98 * 4);
  size_t o_y   = o_ao  + al(S_MC2);
  size_t o_x1  = o_y   + al(S_MC4);
  size_t o_x1h = o_x1  + al(S_MC4);
  // aliases (regions dead by the time these are written):
  size_t o_h   = o_qkv;   // fc1 out (154.1 MB) over qkv+tbl+rpb+attnout (170.9 MB)
  size_t o_hh  = o_y;     // fc2 out over proj-GEMM output

  _Float16* xw_h  = (_Float16*)(ws + o_xw);
  _Float16* wq_h  = (_Float16*)(ws + o_wq);
  _Float16* wp_h  = (_Float16*)(ws + o_wp);
  _Float16* w1_h  = (_Float16*)(ws + o_w1);
  _Float16* w2_h  = (_Float16*)(ws + o_w2);
  _Float16* qkv_h = (_Float16*)(ws + o_qkv);
  float*    tbl   = (float*)(ws + o_tbl);
  float*    rpb   = (float*)(ws + o_rpb);
  _Float16* ao_h  = (_Float16*)(ws + o_ao);
  float*    y     = (float*)(ws + o_y);
  float*    x1    = (float*)(ws + o_x1);
  _Float16* x1_h  = (_Float16*)(ws + o_x1h);
  _Float16* h_h   = (_Float16*)(ws + o_h);
  float*    hh    = (float*)(ws + o_hh);

  const int MT2 = M_ / 32;   // 3136 row-pair tiles

  k_cvtw<<<576, 256, 0, stream>>>(qkv_w, proj_w, fc1_w, fc2_w, wq_h, wp_h, w1_h, w2_h);
  k_window<<<(int)(((size_t)M_ * C_) / 256), 256, 0, stream>>>(x, xw_h);
  k_zero16<<<(int)((S_QKV / 16 + 255) / 256), 256, 0, stream>>>((uint4*)qkv_h,
                                                                (int)(S_QKV / 16));
  k_gemm_qkv<<<dim3(9, MT2), 128, 0, stream>>>(xw_h, wq_h, q_bias, v_bias, qkv_h);
  k_normqk<<<(2 * WHT_ * NP_) / 256, 256, 0, stream>>>(qkv_h, logit_s);
  k_tbl<<<2, 256, 0, stream>>>(cpb_w1, cpb_b1, cpb_w2, tbl);
  k_rpb<<<(NH_ * N_ * N_ + 255) / 256, 256, 0, stream>>>(tbl, rpb);
  k_attn<<<WHT_, 256, 88384, stream>>>(qkv_h, rpb, ao_h);
  k_gemm_proj<<<dim3(3, MT2), 128, 0, stream>>>(ao_h, wp_h, proj_b, y);
  k_unwin_ln1<<<(M_ + 255) / 256, 256, 0, stream>>>(y, x, n1g, n1b, x1, x1_h);
  k_gemm_fc1<<<dim3(12, MT2), 128, 0, stream>>>(x1_h, w1_h, fc1_b, h_h);
  k_gemm_fc2<<<dim3(3, MT2), 128, 0, stream>>>(h_h, w2_h, fc2_b, hh);
  k_final<<<(M_ + 255) / 256, 256, 0, stream>>>(hh, x1, n2g, n2b, (float*)d_out);
}
